// TrajectoryFlowModel_747324309918
// MI455X (gfx1250) — matrix operand order repaired and checked by reading, not compiled, not run on hardware
//
#include <hip/hip_runtime.h>
#include <hip/hip_bf16.h>

typedef __bf16 v16bf __attribute__((ext_vector_type(16)));
typedef float  v8f   __attribute__((ext_vector_type(8)));

#define D_    72
#define MID_  36
#define HID_  64
#define NBLK_ 16

// ---- workspace layout (bytes). One B-fragment = 32 lanes * 16 bf16 = 1024 B ----
#define OFF_LDSUM 0u
#define OFF_XW    1024u
#define SZ_XW     (16u*15u*1024u)          // per blk: 3 ksteps x 5 ntiles
#define OFF_W0    (OFF_XW + SZ_XW)
#define SZ_W0     (16u*8u*1024u)           // per blk: 2 ksteps x 4 ntiles
#define OFF_CW    (OFF_W0 + SZ_W0)
#define SZ_CW     (16u*8u*1024u)           // per blk: 2 gates x 1 kstep x 4 ntiles
#define OFF_L1    (OFF_CW + SZ_CW)
#define SZ_L1     (16u*16u*1024u)          // per blk: 2 gates x 2 ksteps x 4 ntiles
#define OFF_L2    (OFF_L1 + SZ_L1)
#define SZ_L2     (16u*16u*1024u)
#define OFF_WF    (OFF_L2 + SZ_L2)
#define SZ_WF     (16u*10u*1024u)          // per blk: 2 ksteps x 5 ntiles
#define OFF_E1    (OFF_WF + SZ_WF)
#define SZ_E1     (4u*1024u)               // 1 kstep x 4 ntiles
#define OFF_E2    (OFF_E1 + SZ_E1)
#define SZ_E2     (18u*1024u)              // 2 ksteps x 9 ntiles

// =====================================================================
// Prep: build per-block LU product, log-det scalar, and pre-swizzle all
// weights into the WMMA B-fragment lane layout (bf16).
//   B fragment element: lane L, slot j  ->  K = kstep*32 + (L>>4)*16 + j
//                                           N = ntile*16 + (L&15)
// =====================================================================
__global__ void flow_prep_kernel(const float* __restrict__ lL,
                                 const float* __restrict__ lU,
                                 const float* __restrict__ ldg,
                                 const float* __restrict__ W0,
                                 const float* __restrict__ cW,
                                 const float* __restrict__ l1W,
                                 const float* __restrict__ l2W,
                                 const float* __restrict__ Wf,
                                 const float* __restrict__ eW1,
                                 const float* __restrict__ eW2,
                                 unsigned char* __restrict__ ws) {
  const int blk = blockIdx.x;
  const int tid = threadIdx.x;
  __shared__ float sdiag[D_];
  __shared__ float sW[D_][D_];

  for (int k = tid; k < D_; k += blockDim.x)
    sdiag[k] = log1pf(__expf(ldg[blk * D_ + k])) + 0.001f;   // softplus + eps
  __syncthreads();

  if (tid == 0) {
    float s = 0.f;
    for (int k = 0; k < D_; ++k) s += __logf(sdiag[k]);
    ((float*)(ws + OFF_LDSUM))[blk] = s;
  }

  // W = (tril(L,-1)+I) @ (triu(U,1)+diag)
  const float* Lb = lL + blk * D_ * D_;
  const float* Ub = lU + blk * D_ * D_;
  for (int idx = tid; idx < D_ * D_; idx += blockDim.x) {
    int i = idx / D_, k = idx % D_;
    int jm = i < k ? i : k;
    float acc = 0.f;
    for (int j = 0; j <= jm; ++j) {
      float lv = (j < i) ? Lb[i * D_ + j] : 1.0f;
      float uv = (j < k) ? Ub[j * D_ + k] : sdiag[k];
      acc += lv * uv;
    }
    sW[i][k] = acc;
  }
  __syncthreads();

  __bf16* p;
  // x-GEMM weights: B[k][n] = W[n][k], K padded to 96, N padded to 80
  p = (__bf16*)(ws + OFF_XW) + blk * 15 * 512;
  for (int e = tid; e < 15 * 512; e += blockDim.x) {
    int f = e >> 9, r = e & 511, lane = r >> 4, j = r & 15;
    int ks = f / 5, nt = f % 5;
    int k = ks * 32 + (lane >> 4) * 16 + j;
    int n = nt * 16 + (lane & 15);
    p[e] = (__bf16)((k < D_ && n < D_) ? sW[n][k] : 0.f);
  }
  // W0: (64,44), K padded 44->64
  p = (__bf16*)(ws + OFF_W0) + blk * 8 * 512;
  {
    const float* W0b = W0 + blk * HID_ * 44;
    for (int e = tid; e < 8 * 512; e += blockDim.x) {
      int f = e >> 9, r = e & 511, lane = r >> 4, j = r & 15;
      int ks = f >> 2, nt = f & 3;
      int k = ks * 32 + (lane >> 4) * 16 + j;
      int n = nt * 16 + (lane & 15);
      p[e] = (__bf16)((k < 44) ? W0b[n * 44 + k] : 0.f);
    }
  }
  // ctx gate weights: (2,64,8), K padded 8->32
  p = (__bf16*)(ws + OFF_CW) + blk * 8 * 512;
  for (int e = tid; e < 8 * 512; e += blockDim.x) {
    int f = e >> 9, r = e & 511, lane = r >> 4, j = r & 15;
    int gi = f >> 2, nt = f & 3;
    int k = (lane >> 4) * 16 + j;
    int n = nt * 16 + (lane & 15);
    p[e] = (__bf16)((k < 8) ? cW[((blk * 2 + gi) * HID_ + n) * 8 + k] : 0.f);
  }
  // l1W / l2W: (2,64,64)
  p = (__bf16*)(ws + OFF_L1) + blk * 16 * 512;
  for (int e = tid; e < 16 * 512; e += blockDim.x) {
    int f = e >> 9, r = e & 511, lane = r >> 4, j = r & 15;
    int gi = f >> 3, ks = (f >> 2) & 1, nt = f & 3;
    int k = ks * 32 + (lane >> 4) * 16 + j;
    int n = nt * 16 + (lane & 15);
    p[e] = (__bf16)l1W[((blk * 2 + gi) * HID_ + n) * HID_ + k];
  }
  p = (__bf16*)(ws + OFF_L2) + blk * 16 * 512;
  for (int e = tid; e < 16 * 512; e += blockDim.x) {
    int f = e >> 9, r = e & 511, lane = r >> 4, j = r & 15;
    int gi = f >> 3, ks = (f >> 2) & 1, nt = f & 3;
    int k = ks * 32 + (lane >> 4) * 16 + j;
    int n = nt * 16 + (lane & 15);
    p[e] = (__bf16)l2W[((blk * 2 + gi) * HID_ + n) * HID_ + k];
  }
  // Wf: (72,64), N padded 72->80
  p = (__bf16*)(ws + OFF_WF) + blk * 10 * 512;
  for (int e = tid; e < 10 * 512; e += blockDim.x) {
    int f = e >> 9, r = e & 511, lane = r >> 4, j = r & 15;
    int ks = f / 5, nt = f % 5;
    int k = ks * 32 + (lane >> 4) * 16 + j;
    int n = nt * 16 + (lane & 15);
    p[e] = (__bf16)((n < D_) ? Wf[(blk * D_ + n) * HID_ + k] : 0.f);
  }
  // encoder (only block 0)
  if (blk == 0) {
    p = (__bf16*)(ws + OFF_E1);
    for (int e = tid; e < 4 * 512; e += blockDim.x) {
      int f = e >> 9, r = e & 511, lane = r >> 4, j = r & 15;
      int k = (lane >> 4) * 16 + j;
      int n = f * 16 + (lane & 15);
      p[e] = (__bf16)((k < 8) ? eW1[n * 8 + k] : 0.f);
    }
    p = (__bf16*)(ws + OFF_E2);
    for (int e = tid; e < 18 * 512; e += blockDim.x) {
      int f = e >> 9, r = e & 511, lane = r >> 4, j = r & 15;
      int ks = f / 9, nt = f % 9;
      int k = ks * 32 + (lane >> 4) * 16 + j;
      int n = nt * 16 + (lane & 15);
      p[e] = (__bf16)eW2[n * HID_ + k];
    }
  }
}

// =====================================================================
// WMMA helpers
// =====================================================================
__device__ __forceinline__ v8f wmma_bf(v16bf a, v16bf b, v8f c) {
  return __builtin_amdgcn_wmma_f32_16x16x32_bf16(false, a, false, b, (short)0, c,
                                                 false, false);
}
// A fragment from an LDS f32 tile (row-major, given stride), contiguous k order (same as B packing):
//   lane L: M = L&15;  slot j -> K = kbase + (L>>4)*16 + j
__device__ __forceinline__ v16bf mkA(const float* row, int kbase, int hi, int relu) {
  v16bf a;
#pragma unroll
  for (int j = 0; j < 16; ++j) {
    int k = kbase + hi * 16 + j;
    float v = row[k];
    if (relu) v = fmaxf(v, 0.f);
    a[j] = (__bf16)v;
  }
  return a;
}
__device__ __forceinline__ v16bf ldB(const __bf16* fr, int lane) {
  return *(const v16bf*)(fr + lane * 16);   // 32B contiguous per lane
}
__device__ __forceinline__ v8f biasC(const float* b, int n, int nmax) {
  float bv = (n < nmax) ? b[n] : 0.f;
  v8f c;
#pragma unroll
  for (int r = 0; r < 8; ++r) c[r] = bv;
  return c;
}
// C/D layout: VGPR r -> M = mtbase + hi*8 + r, N = ncol
__device__ __forceinline__ void stC(float* buf, int stride, int mtbase, int hi,
                                    int ncol, v8f c) {
#pragma unroll
  for (int r = 0; r < 8; ++r) buf[(mtbase + hi * 8 + r) * stride + ncol] = c[r];
}

// =====================================================================
// Main flow kernel: one wave32 per 32 batch rows.
// =====================================================================
__launch_bounds__(32)
__global__ void flow_main_kernel(const float* __restrict__ traj,
                                 const float* __restrict__ start,
                                 const float* __restrict__ goal,
                                 const float* __restrict__ lu_bias,
                                 const float* __restrict__ b0,
                                 const float* __restrict__ ctxb,
                                 const float* __restrict__ l1b,
                                 const float* __restrict__ l2b,
                                 const float* __restrict__ bfb,
                                 const float* __restrict__ eb1,
                                 const float* __restrict__ eb2,
                                 const unsigned char* __restrict__ ws,
                                 float* __restrict__ out) {
  const int lane = threadIdx.x;
  const int m16 = lane & 15, hi = lane >> 4;
  const long base = (long)blockIdx.x * 32;

  __shared__ __align__(16) float xs[32][96];   // x, K-padded to 96 (cols 72..95 = 0)
  __shared__ __align__(16) float ts[32][64];   // resnet hidden t
  __shared__ __align__(16) float gs[2][32][64];// gates
  __shared__ __align__(16) float ps[32][144];  // scratch a / params / encoder p
  __shared__ __align__(16) float cs[32][8];    // ctx

  // ---- load x tile (32 rows x 72 f32) ----
  for (int i = lane; i < 32 * 18; i += 32) {
    int m = i / 18, q = i % 18;
    ((float4*)&xs[m][0])[q] = ((const float4*)(traj + (base + m) * D_))[q];
  }
  {
    float4 z4 = make_float4(0.f, 0.f, 0.f, 0.f);
    for (int i = lane; i < 32 * 6; i += 32) {
      int m = i / 6, q = i % 6;
      ((float4*)&xs[m][72])[q] = z4;
    }
  }
  // ---- ctx = [start, goal] ----
  for (int i = lane; i < 32 * 2; i += 32) {
    int m = i >> 1, h = i & 1;
    ((float4*)&cs[m][0])[h] = ((const float4*)((h ? goal : start) + (base + m) * 4))[0];
  }

  // ctx A fragments (K padded 8->32) — reused by every gate GEMM and the encoder
  v16bf actx[2];
#pragma unroll
  for (int mt = 0; mt < 2; ++mt) {
    v16bf a;
#pragma unroll
    for (int j = 0; j < 16; ++j) {
      int k = hi * 16 + j;
      a[j] = (__bf16)((k < 8) ? cs[mt * 16 + m16][k] : 0.f);
    }
    actx[mt] = a;
  }

  float ldacc = 0.f;  // log-det accumulator for row m = lane
  const float* ldsum = (const float*)(ws + OFF_LDSUM);

  for (int blk = 0; blk < NBLK_; ++blk) {
    // ============ x = x @ W^T + lb  (M=32, N=80, K=96) ============
    const __bf16* fW = (const __bf16*)(ws + OFF_XW) + blk * 15 * 512;
    v16bf ax[2][3];
#pragma unroll
    for (int mt = 0; mt < 2; ++mt)
#pragma unroll
      for (int ks = 0; ks < 3; ++ks)
        ax[mt][ks] = mkA(&xs[mt * 16 + m16][0], ks * 32, hi, 0);
    const float* lb = lu_bias + blk * D_;
#pragma unroll
    for (int nt = 0; nt < 5; ++nt) {
      int n = nt * 16 + m16;
      v8f c0 = biasC(lb, n, D_), c1 = c0;
#pragma unroll
      for (int ks = 0; ks < 3; ++ks) {
        v16bf b = ldB(fW + (ks * 5 + nt) * 512, lane);
        c0 = wmma_bf(ax[0][ks], b, c0);
        c1 = wmma_bf(ax[1][ks], b, c1);
      }
      stC(&xs[0][0], 96, 0, hi, n, c0);
      stC(&xs[0][0], 96, 16, hi, n, c1);
    }
    ldacc += ldsum[blk];

    // ============ t = [h_id, ctx] @ W0^T + b0  (K=44->64) ============
    const __bf16* fW0 = (const __bf16*)(ws + OFF_W0) + blk * 8 * 512;
    v16bf ah[2][2];
#pragma unroll
    for (int mt = 0; mt < 2; ++mt)
#pragma unroll
      for (int ks = 0; ks < 2; ++ks) {
        v16bf a;
        int m = mt * 16 + m16;
#pragma unroll
        for (int j = 0; j < 16; ++j) {
          int k = ks * 32 + hi * 16 + j;
          float v = 0.f;
          if (k < 36)      v = xs[m][36 + k];
          else if (k < 44) v = cs[m][k - 36];
          a[j] = (__bf16)v;
        }
        ah[mt][ks] = a;
      }
    const float* b0b = b0 + blk * HID_;
#pragma unroll
    for (int nt = 0; nt < 4; ++nt) {
      int n = nt * 16 + m16;
      v8f c0 = biasC(b0b, n, HID_), c1 = c0;
#pragma unroll
      for (int ks = 0; ks < 2; ++ks) {
        v16bf b = ldB(fW0 + (ks * 4 + nt) * 512, lane);
        c0 = wmma_bf(ah[0][ks], b, c0);
        c1 = wmma_bf(ah[1][ks], b, c1);
      }
      stC(&ts[0][0], 64, 0, hi, n, c0);
      stC(&ts[0][0], 64, 16, hi, n, c1);
    }

    // ============ gates g[i] = sigmoid(ctx @ cW^T + cb) ============
    const __bf16* fcW = (const __bf16*)(ws + OFF_CW) + blk * 8 * 512;
#pragma unroll
    for (int gi = 0; gi < 2; ++gi) {
      const float* cbb = ctxb + (blk * 2 + gi) * HID_;
#pragma unroll
      for (int nt = 0; nt < 4; ++nt) {
        int n = nt * 16 + m16;
        v16bf b = ldB(fcW + (gi * 4 + nt) * 512, lane);
        v8f c0 = biasC(cbb, n, HID_), c1 = c0;
        c0 = wmma_bf(actx[0], b, c0);
        c1 = wmma_bf(actx[1], b, c1);
#pragma unroll
        for (int r = 0; r < 8; ++r) {
          gs[gi][hi * 8 + r][n]      = 1.f / (1.f + __expf(-c0[r]));
          gs[gi][16 + hi * 8 + r][n] = 1.f / (1.f + __expf(-c1[r]));
        }
      }
    }

    // ============ two residual rounds ============
#pragma unroll
    for (int gi = 0; gi < 2; ++gi) {
      // a = relu(t) @ l1W^T + l1b  -> ps[:, :64]
      const __bf16* f1 = (const __bf16*)(ws + OFF_L1) + (blk * 16 + gi * 8) * 512;
      v16bf at[2][2];
#pragma unroll
      for (int mt = 0; mt < 2; ++mt)
#pragma unroll
        for (int ks = 0; ks < 2; ++ks)
          at[mt][ks] = mkA(&ts[mt * 16 + m16][0], ks * 32, hi, 1);
      const float* b1 = l1b + (blk * 2 + gi) * HID_;
#pragma unroll
      for (int nt = 0; nt < 4; ++nt) {
        int n = nt * 16 + m16;
        v8f c0 = biasC(b1, n, HID_), c1 = c0;
#pragma unroll
        for (int ks = 0; ks < 2; ++ks) {
          v16bf b = ldB(f1 + (ks * 4 + nt) * 512, lane);
          c0 = wmma_bf(at[0][ks], b, c0);
          c1 = wmma_bf(at[1][ks], b, c1);
        }
        stC(&ps[0][0], 144, 0, hi, n, c0);
        stC(&ps[0][0], 144, 16, hi, n, c1);
      }
      // a2 = relu(a) @ l2W^T + l2b ; t += a2 * g
      const __bf16* f2 = (const __bf16*)(ws + OFF_L2) + (blk * 16 + gi * 8) * 512;
      v16bf aa[2][2];
#pragma unroll
      for (int mt = 0; mt < 2; ++mt)
#pragma unroll
        for (int ks = 0; ks < 2; ++ks)
          aa[mt][ks] = mkA(&ps[mt * 16 + m16][0], ks * 32, hi, 1);
      const float* b2 = l2b + (blk * 2 + gi) * HID_;
#pragma unroll
      for (int nt = 0; nt < 4; ++nt) {
        int n = nt * 16 + m16;
        v8f c0 = biasC(b2, n, HID_), c1 = c0;
#pragma unroll
        for (int ks = 0; ks < 2; ++ks) {
          v16bf b = ldB(f2 + (ks * 4 + nt) * 512, lane);
          c0 = wmma_bf(aa[0][ks], b, c0);
          c1 = wmma_bf(aa[1][ks], b, c1);
        }
#pragma unroll
        for (int r = 0; r < 8; ++r) {
          int mA = hi * 8 + r, mB = 16 + hi * 8 + r;
          ts[mA][n] += c0[r] * gs[gi][mA][n];
          ts[mB][n] += c1[r] * gs[gi][mB][n];
        }
      }
    }

    // ============ params = t @ Wf^T + bf  (N=72->80) ============
    const __bf16* fWf = (const __bf16*)(ws + OFF_WF) + blk * 10 * 512;
    v16bf atf[2][2];
#pragma unroll
    for (int mt = 0; mt < 2; ++mt)
#pragma unroll
      for (int ks = 0; ks < 2; ++ks)
        atf[mt][ks] = mkA(&ts[mt * 16 + m16][0], ks * 32, hi, 0);
    const float* bff = bfb + blk * D_;
#pragma unroll
    for (int nt = 0; nt < 5; ++nt) {
      int n = nt * 16 + m16;
      v8f c0 = biasC(bff, n, D_), c1 = c0;
#pragma unroll
      for (int ks = 0; ks < 2; ++ks) {
        v16bf b = ldB(fWf + (ks * 5 + nt) * 512, lane);
        c0 = wmma_bf(atf[0][ks], b, c0);
        c1 = wmma_bf(atf[1][ks], b, c1);
      }
      stC(&ps[0][0], 144, 0, hi, n, c0);
      stC(&ps[0][0], 144, 16, hi, n, c1);
    }

    // ============ affine update + logdet (row m = lane) ============
    {
      const int m = lane;
#pragma unroll 4
      for (int j = 0; j < MID_; ++j) {
        float sc = 1.f / (1.f + __expf(-(ps[m][MID_ + j] + 2.f))) + 0.001f;
        xs[m][j] = xs[m][j] * sc + ps[m][j];
        ldacc += __logf(sc);
      }
    }
  }  // blocks

  // ============ encoder: h = relu(ctx@W1^T+b1); p = h@W2^T+b2 ============
  const __bf16* fe1 = (const __bf16*)(ws + OFF_E1);
#pragma unroll
  for (int nt = 0; nt < 4; ++nt) {
    int n = nt * 16 + m16;
    v16bf b = ldB(fe1 + nt * 512, lane);
    v8f c0 = biasC(eb1, n, HID_), c1 = c0;
    c0 = wmma_bf(actx[0], b, c0);
    c1 = wmma_bf(actx[1], b, c1);
    stC(&ts[0][0], 64, 0, hi, n, c0);
    stC(&ts[0][0], 64, 16, hi, n, c1);
  }
  const __bf16* fe2 = (const __bf16*)(ws + OFF_E2);
  v16bf ae[2][2];
#pragma unroll
  for (int mt = 0; mt < 2; ++mt)
#pragma unroll
    for (int ks = 0; ks < 2; ++ks)
      ae[mt][ks] = mkA(&ts[mt * 16 + m16][0], ks * 32, hi, 1);
#pragma unroll
  for (int nt = 0; nt < 9; ++nt) {
    int n = nt * 16 + m16;
    v8f c0 = biasC(eb2, n, 2 * D_), c1 = c0;
#pragma unroll
    for (int ks = 0; ks < 2; ++ks) {
      v16bf b = ldB(fe2 + (ks * 9 + nt) * 512, lane);
      c0 = wmma_bf(ae[0][ks], b, c0);
      c1 = wmma_bf(ae[1][ks], b, c1);
    }
    stC(&ps[0][0], 144, 0, hi, n, c0);
    stC(&ps[0][0], 144, 16, hi, n, c1);
  }

  // ============ base log-prob (row m = lane) ============
  {
    const int m = lane;
    float acc = 0.f;
#pragma unroll 4
    for (int j = 0; j < D_; ++j) {
      float ls = ps[m][D_ + j];
      float z = (xs[m][j] - ps[m][j]) * __expf(-ls);
      acc += -0.5f * z * z - ls;
    }
    out[base + m] = acc + ldacc - 66.16357439f;  // 0.5*D*log(2*pi)
  }
}

extern "C" void kernel_launch(void* const* d_in, const int* in_sizes, int n_in,
                              void* d_out, int out_size, void* d_ws, size_t ws_size,
                              hipStream_t stream) {
  const float* traj  = (const float*)d_in[0];
  const float* start = (const float*)d_in[1];
  const float* goal  = (const float*)d_in[2];
  const float* lL    = (const float*)d_in[3];
  const float* lU    = (const float*)d_in[4];
  const float* ldg   = (const float*)d_in[5];
  const float* lb    = (const float*)d_in[6];
  const float* W0    = (const float*)d_in[7];
  const float* b0    = (const float*)d_in[8];
  const float* cW    = (const float*)d_in[9];
  const float* cb    = (const float*)d_in[10];
  const float* l1W   = (const float*)d_in[11];
  const float* l1b   = (const float*)d_in[12];
  const float* l2W   = (const float*)d_in[13];
  const float* l2b   = (const float*)d_in[14];
  const float* Wf    = (const float*)d_in[15];
  const float* bf    = (const float*)d_in[16];
  const float* eW1   = (const float*)d_in[17];
  const float* eb1   = (const float*)d_in[18];
  const float* eW2   = (const float*)d_in[19];
  const float* eb2   = (const float*)d_in[20];
  unsigned char* ws  = (unsigned char*)d_ws;

  flow_prep_kernel<<<NBLK_, 256, 0, stream>>>(lL, lU, ldg, W0, cW, l1W, l2W, Wf,
                                              eW1, eW2, ws);

  int Bn = in_sizes[0] / D_;  // 65536
  flow_main_kernel<<<Bn / 32, 32, 0, stream>>>(traj, start, goal, lb, b0, cb, l1b,
                                               l2b, bf, eb1, eb2, ws, (float*)d_out);
}
